// IlluminationGuidedMSA_55929064129346
// MI455X (gfx1250) — matrix operand order repaired and checked by reading, not compiled, not run on hardware
//
#include <hip/hip_runtime.h>
#include <hip/hip_fp16.h>

// ---------------------------------------------------------------------------
// IlluminationGuidedMSA for MI455X (gfx1250, wave32, WMMA)
//   b=2, c=64, h=w=64  ->  bh = b*heads = 8, N = 4096, d = 16
//   attention done in f16 WMMA (v_wmma_f32_16x16x32_f16) with fp32 softmax
//   in the exp2 domain (log2(e) folded into the Q scale).
// ---------------------------------------------------------------------------

typedef __attribute__((ext_vector_type(16))) _Float16 v16h;
typedef __attribute__((ext_vector_type(8)))  float    v8f;
typedef __attribute__((ext_vector_type(4)))  float    v4f;

#define HWN   4096      // h*w
#define CCH   64        // channels
#define HEADS 4
#define DD    16        // head dim
#define NB    2         // batch
#define BH    8         // batch*heads

// scale = d^-0.5 * log2(e) so the softmax can use raw v_exp_f32 (exp2)
#define QSCALE 0.36067376022224085f

// ---------------------------------------------------------------------------
// Kernel 1: per-pixel 1x1 convs (qkv + illu). Produces
//   Qh [bh][N][16] f16  (pre-scaled by d^-0.5 * log2e)
//   Kh [bh][N][16] f16  (k + illu)
//   Vt [bh][16][N] f16  (V transposed so PV A-operand loads are contiguous)
// grid = (32, 12): x = (b, pixel-tile of 256), y = 16-channel chunk of qkv.
// ---------------------------------------------------------------------------
__global__ __launch_bounds__(256) void qkv_illu_kernel(
    const float* __restrict__ x, const float* __restrict__ illu,
    const float* __restrict__ qkv_w, const float* __restrict__ qkv_b,
    const float* __restrict__ illu_w, const float* __restrict__ illu_b,
    _Float16* __restrict__ Qh, _Float16* __restrict__ Kh,
    _Float16* __restrict__ Vt)
{
    const int t = threadIdx.x;
    const int b = blockIdx.x >> 4;                    // 16 pixel-tiles / batch
    const int n = ((blockIdx.x & 15) << 8) + t;       // pixel index in [0,4096)
    const int chan0 = blockIdx.y * 16;                // 0..176, 16-aligned
    const int three = chan0 >> 6;                     // 0=q 1=k 2=v
    const int head  = (chan0 >> 4) & 3;

    const float* xb = x + (size_t)b * CCH * HWN;
    float xv[CCH];
#pragma unroll
    for (int c = 0; c < CCH; ++c) xv[c] = xb[c * HWN + n];   // coalesced

    const bool isK = (three == 1);
    float iv[CCH];
    if (isK) {
        const float* ib = illu + (size_t)b * CCH * HWN;
#pragma unroll
        for (int c = 0; c < CCH; ++c) iv[c] = ib[c * HWN + n];
    }

    const size_t bh = (size_t)b * HEADS + head;
    for (int j = 0; j < 16; ++j) {
        const int co = chan0 + j;         // qkv output channel (0..191)
        const int cc = co & 63;           // (head,d) channel within third
        const float* w = qkv_w + co * CCH;
        float acc = qkv_b[co];
#pragma unroll
        for (int c = 0; c < CCH; ++c) acc += xv[c] * w[c];
        if (isK) {
            const float* wi = illu_w + cc * CCH;
            float a2 = illu_b[cc];
#pragma unroll
            for (int c = 0; c < CCH; ++c) a2 += iv[c] * wi[c];
            acc += a2;
        }
        const int d = j;                  // chan0 is 16-aligned -> d = cc & 15
        if (three == 0)
            Qh[(bh * HWN + n) * DD + d] = (_Float16)(acc * QSCALE);
        else if (three == 1)
            Kh[(bh * HWN + n) * DD + d] = (_Float16)acc;
        else
            Vt[(bh * DD + d) * HWN + n] = (_Float16)acc;   // coalesced
    }
}

// ---------------------------------------------------------------------------
// Kernel 2: flash attention. grid = (N/64, BH), block = 128 (4 waves).
// Each wave: 16 query rows, streams keys 32 at a time.
//   S^T = K_tile(16x32, d-padded) x Q^T(32x16)   -> key dim lands in the M /
//   VGPR direction of the C/D layout, so online-softmax is lane-local plus one
//   xor-16 shuffle, and P^T packing / V^T loads are lane-local & contiguous.
// The B operand (Q^T) carries hard zeros at K=16..31 (lanes 16-31), so the
// A operands' K=16..31 elements are multiplied by zero: we load them as one
// contiguous 32B v16h with finite "don't care" data in the upper half -> no
// zero-padding register traffic in the loop.
// ---------------------------------------------------------------------------
__global__ __launch_bounds__(128) void attn_kernel(
    const _Float16* __restrict__ Qh, const _Float16* __restrict__ Kh,
    const _Float16* __restrict__ Vt, float* __restrict__ Of)
{
    const int lane  = threadIdx.x & 31;
    const int wave  = threadIdx.x >> 5;
    const int bh    = blockIdx.y;
    const int qbase = blockIdx.x * 64 + wave * 16;
    const int half  = lane >> 4;     // 0: lanes 0-15, 1: lanes 16-31
    const int l16   = lane & 15;

    // B-operand Q^T (32x16, d padded to 32): lane<16 holds K=0..15 of its
    // column (full 16-d query row, one 32B load); lane>=16 holds the zero pad.
    const _Float16* qp = Qh + ((size_t)bh * HWN + qbase + l16) * DD;
    v16h bz = {};
    v16h bqv = *(const v16h*)qp;
    v16h bq  = (half == 0) ? bqv : bz;

    const _Float16* kp = Kh + (size_t)bh * HWN * DD;
    const _Float16* vp = Vt + ((size_t)bh * DD + l16) * HWN;  // row d = l16

    float m = -1e30f, l = 0.f;
    v8f acc = {};
    const v8f zero8 = {};

    for (int kb = 0; kb < HWN; kb += 32) {
        if (kb + 64 < HWN) {  // prefetch next-next tile (global_prefetch_b8)
            __builtin_prefetch(kp + (kb + 64 + l16) * DD, 0, 1);
            __builtin_prefetch(vp + kb + 64, 0, 1);
        }
        // A-operands: K rows (16x32). lane<16 needs d0..15 in elems 0..15,
        // lane>=16 elems are all don't-care (B is zero there),
        // so a single contiguous 32B load suffices.
        v16h ak0 = *(const v16h*)(kp + (size_t)(kb + l16) * DD + half * 8);
        v16h ak1 = *(const v16h*)(kp + (size_t)(kb + 16 + l16) * DD + half * 8);

        v8f s0 = __builtin_amdgcn_wmma_f32_16x16x32_f16(
            false, ak0, false, bq, (short)0, zero8, false, false);
        v8f s1 = __builtin_amdgcn_wmma_f32_16x16x32_f16(
            false, ak1, false, bq, (short)0, zero8, false, false);

        // online softmax (exp2 domain) over this half's 16 keys + partner half
        float tmax = s0[0];
#pragma unroll
        for (int i = 1; i < 8; ++i) tmax = fmaxf(tmax, s0[i]);
#pragma unroll
        for (int i = 0; i < 8; ++i) tmax = fmaxf(tmax, s1[i]);
        tmax = fmaxf(tmax, __shfl_xor(tmax, 16, 32));
        const float nm = fmaxf(m, tmax);
        const float alpha = __builtin_amdgcn_exp2f(m - nm);

        float lsum = 0.f;
        v16h bp;   // P^T B-operand: elems 0..7 <- S0 exps, 8..15 <- S1 exps
#pragma unroll
        for (int i = 0; i < 8; ++i) {
            float e = __builtin_amdgcn_exp2f(s0[i] - nm);
            lsum += e; bp[i] = (_Float16)e;
        }
#pragma unroll
        for (int i = 0; i < 8; ++i) {
            float e = __builtin_amdgcn_exp2f(s1[i] - nm);
            lsum += e; bp[8 + i] = (_Float16)e;
        }
        lsum += __shfl_xor(lsum, 16, 32);
        l = l * alpha + lsum;
        m = nm;
#pragma unroll
        for (int i = 0; i < 8; ++i) acc[i] *= alpha;

        // A-operand V^T (16x32 keys), matching bp: elems 0..7 = keys
        // kb+8*half..+7, elems 8..15 = keys kb+16+8*half..+7 of row d = l16.
        v16h av;
        const _Float16* vq = vp + kb + half * 8;
#pragma unroll
        for (int i = 0; i < 8; ++i) { av[i] = vq[i]; av[8 + i] = vq[16 + i]; }

        acc = __builtin_amdgcn_wmma_f32_16x16x32_f16(
            false, av, false, bp, (short)0, acc, false, false);
    }

    // acc = O^T (d x query), lane owns query column qbase+l16, rows half*8..+7
    const float inv = 1.f / l;
    float* op = Of + ((size_t)bh * HWN + qbase + l16) * DD + half * 8;
    v4f o0 = { acc[0]*inv, acc[1]*inv, acc[2]*inv, acc[3]*inv };
    v4f o1 = { acc[4]*inv, acc[5]*inv, acc[6]*inv, acc[7]*inv };
    *(v4f*)op       = o0;
    *((v4f*)op + 1) = o1;
}

// ---------------------------------------------------------------------------
// Kernel 3: output projection y[b][co][n] = Of[b][:, n] . proj_w[co] + proj_b.
// grid = (N/256, b*4 co-groups), 16 output channels per thread.
// ---------------------------------------------------------------------------
__global__ __launch_bounds__(256) void proj_kernel(
    const float* __restrict__ Of, const float* __restrict__ proj_w,
    const float* __restrict__ proj_b, float* __restrict__ y)
{
    const int t = threadIdx.x;
    const int n = blockIdx.x * 256 + t;
    const int g = blockIdx.y;
    const int b  = g >> 2;
    const int cg = (g & 3) * 16;

    float ofv[64];
#pragma unroll
    for (int h = 0; h < 4; ++h) {
        const float* p = Of + ((size_t)(b * HEADS + h) * HWN + n) * DD;
#pragma unroll
        for (int d = 0; d < 16; ++d) ofv[h * 16 + d] = p[d];
    }
    for (int j = 0; j < 16; ++j) {
        const int co = cg + j;
        const float* w = proj_w + co * CCH;
        float acc = proj_b[co];
#pragma unroll
        for (int c = 0; c < CCH; ++c) acc += ofv[c] * w[c];
        y[((size_t)b * CCH + co) * HWN + n] = acc;    // coalesced
    }
}

// ---------------------------------------------------------------------------
extern "C" void kernel_launch(void* const* d_in, const int* in_sizes, int n_in,
                              void* d_out, int out_size, void* d_ws,
                              size_t ws_size, hipStream_t stream)
{
    (void)in_sizes; (void)n_in; (void)out_size; (void)ws_size;
    const float* x      = (const float*)d_in[0];
    const float* illu   = (const float*)d_in[1];
    const float* qkv_w  = (const float*)d_in[2];
    const float* qkv_b  = (const float*)d_in[3];
    const float* illu_w = (const float*)d_in[4];
    const float* illu_b = (const float*)d_in[5];
    const float* proj_w = (const float*)d_in[6];
    const float* proj_b = (const float*)d_in[7];
    float* y = (float*)d_out;

    // workspace carve-up (5 MB total, all offsets 1 MB aligned)
    char* ws = (char*)d_ws;
    _Float16* Qh = (_Float16*)(ws);                 // 1 MB
    _Float16* Kh = (_Float16*)(ws + (1u << 20));    // 1 MB
    _Float16* Vt = (_Float16*)(ws + (2u << 20));    // 1 MB
    float*    Of = (float*)   (ws + (3u << 20));    // 2 MB

    qkv_illu_kernel<<<dim3(16 * NB, 12), dim3(256), 0, stream>>>(
        x, illu, qkv_w, qkv_b, illu_w, illu_b, Qh, Kh, Vt);
    attn_kernel<<<dim3(HWN / 64, BH), dim3(128), 0, stream>>>(Qh, Kh, Vt, Of);
    proj_kernel<<<dim3(HWN / 256, NB * 4), dim3(256), 0, stream>>>(
        Of, proj_w, proj_b, y);
}
